// FeedForwardSNN_4604204941428
// MI455X (gfx1250) — matrix operand order repaired and checked by reading, not compiled, not run on hardware
//
#include <hip/hip_runtime.h>
#include <hip/hip_bf16.h>

typedef __attribute__((ext_vector_type(16))) _Float16 v16h;
typedef __attribute__((ext_vector_type(8)))  _Float16 v8h;
typedef __attribute__((ext_vector_type(8)))  float    v8f;

#define TT   500
#define BB   256
#define DIN  700
#define KP   704      // DIN padded to multiple of 32
#define HH   512
#define DOUT 20

// workspace element offsets (in _Float16 units)
#define OFF_WIN 0
#define OFF_W1  (512*704)               // 360448
#define OFF_W2  (OFF_W1 + 512*512)      // 622592
#define OFF_X0  (OFF_W2 + 32*512)       // 638976  (then 500*256*512 halves)

// Build a 16x32 f16 A-fragment in contiguous K order: lane<16 -> row K[0..16),
// lane>=16 -> row K[16..32). `row` is the base of this lane's A row.
__device__ __forceinline__ v16h load_afrag(const _Float16* row, int kt, int half) {
    const _Float16* base = row + kt * 32 + half * 16;
    v8h lo = *(const v8h*)(base);
    v8h hi = *(const v8h*)(base + 8);
    return __builtin_shufflevector(lo, hi, 0,1,2,3,4,5,6,7,8,9,10,11,12,13,14,15);
}

// ---------------- kernel 0: convert weights to f16 (with padding) ----------------
__global__ __launch_bounds__(256) void cvt_weights_kernel(const float* __restrict__ Win,
                                                          const float* __restrict__ W1,
                                                          const float* __restrict__ W2,
                                                          _Float16* __restrict__ ws) {
    int i = blockIdx.x * 256 + threadIdx.x;
    if (i < OFF_W1) {                       // W_in: [512][704], K padded with 0
        int r = i / KP, k = i % KP;
        ws[i] = (k < DIN) ? (_Float16)Win[r * DIN + k] : (_Float16)0.0f;
    } else if (i < OFF_W2) {                // W1: [512][512]
        ws[i] = (_Float16)W1[i - OFF_W1];
    } else if (i < OFF_X0) {                // W2: [32][512], rows 20..31 = 0
        int j = i - OFF_W2;
        int r = j / HH, k = j % HH;
        ws[i] = (r < DOUT) ? (_Float16)W2[r * HH + k] : (_Float16)0.0f;
    }
}

// ---------------- kernel 1: X0[t*B+b][h] = inp @ W_in^T + b_in  (f16 out) --------
// grid = 4000 tiles of 32 rows; block = 256 (8 waves); each wave owns 4 N-tiles
// and both 16-row M-tiles -> every B fragment feeds 2 WMMAs (B-traffic halved).
__global__ __launch_bounds__(256) void gemm0_kernel(const float* __restrict__ inp,
                                                    const float* __restrict__ b_in,
                                                    const _Float16* __restrict__ Wf,
                                                    _Float16* __restrict__ X0) {
    __shared__ _Float16 At[32][712];        // 32 rows x 704 (pad 712 for alignment)
    const int tid  = threadIdx.x;
    const int row0 = blockIdx.x * 32;
    {
        const int rr = tid >> 3;            // 32 rows, 8 threads per row
        const int cs = tid & 7;
        const float* src = inp + (size_t)(row0 + rr) * DIN;
        #pragma unroll
        for (int i = 0; i < 88; ++i) {      // 88*8 = 704
            int c = i * 8 + cs;
            At[rr][c] = (c < DIN) ? (_Float16)src[c] : (_Float16)0.0f;
        }
    }
    __syncthreads();

    const int wave = tid >> 5;
    const int lane = tid & 31;
    const int half = lane >> 4;
    const int lc   = lane & 15;

    v8f acc0[4] = {};                       // M-tile 0 (rows 0..15)
    v8f acc1[4] = {};                       // M-tile 1 (rows 16..31)
    const _Float16* arow0 = &At[lc][0];
    const _Float16* arow1 = &At[16 + lc][0];
    const _Float16* bbase = Wf + (size_t)(wave * 64 + lc) * KP + half * 16;
    #pragma unroll 2
    for (int kt = 0; kt < 22; ++kt) {       // K = 704
        v16h a0 = load_afrag(arow0, kt, half);
        v16h a1 = load_afrag(arow1, kt, half);
        #pragma unroll
        for (int j = 0; j < 4; ++j) {
            v16h b = *(const v16h*)(bbase + (size_t)j * 16 * KP + kt * 32);
            acc0[j] = __builtin_amdgcn_wmma_f32_16x16x32_f16(false, a0, false, b,
                                                             (short)0, acc0[j], false, false);
            acc1[j] = __builtin_amdgcn_wmma_f32_16x16x32_f16(false, a1, false, b,
                                                             (short)0, acc1[j], false, false);
        }
    }
    #pragma unroll
    for (int j = 0; j < 4; ++j) {
        int n = wave * 64 + j * 16 + lc;
        float bb = b_in[n];
        #pragma unroll
        for (int i = 0; i < 8; ++i) {
            int m = i + 8 * half;
            X0[(size_t)(row0 + m) * HH + n]      = (_Float16)(acc0[j][i] + bb);
            X0[(size_t)(row0 + 16 + m) * HH + n] = (_Float16)(acc1[j][i] + bb);
        }
    }
}

// ---------------- kernel 2: sequential LIF scan over T ---------------------------
// grid = 16 WGs (16 batch rows each); block = 256 (8 waves). v1/v2/vo live in
// registers with the WMMA C/D mapping; spikes staged in LDS as WMMA A operands.
__global__ __launch_bounds__(256) void scan_kernel(const float* __restrict__ b1g,
                                                   const float* __restrict__ b2g,
                                                   const _Float16* __restrict__ W1f,
                                                   const _Float16* __restrict__ W2f,
                                                   const _Float16* __restrict__ X0,
                                                   float* __restrict__ out) {
    __shared__ _Float16 s1[16][520];
    __shared__ _Float16 s2[16][520];

    const int tid  = threadIdx.x;
    const int wave = tid >> 5;
    const int lane = tid & 31;
    const int half = lane >> 4;
    const int lc   = lane & 15;
    const int wg   = blockIdx.x;

    float v1r[32], v2r[32], vor[8];
    #pragma unroll
    for (int i = 0; i < 32; ++i) { v1r[i] = 0.0f; v2r[i] = 0.0f; }
    #pragma unroll
    for (int i = 0; i < 8; ++i) vor[i] = 0.0f;

    float bb1[4];
    #pragma unroll
    for (int j = 0; j < 4; ++j) bb1[j] = b1g[wave * 64 + j * 16 + lc];
    const int n2 = wave * 16 + lc;          // output column for waves 0,1
    float bb2 = 0.0f;
    if (wave < 2 && n2 < DOUT) bb2 = b2g[n2];

    const _Float16* bbase1 = W1f + (size_t)(wave * 64 + lc) * HH + half * 16;
    const _Float16* bbase2 = W2f + (size_t)n2 * HH + half * 16;
    const _Float16* x0tile = X0 + (size_t)wg * 16 * HH;   // this WG's rows at t=0

    for (int t = 0; t < TT; ++t) {
        // phase 1: v1 = 0.9 v1 + x0 ; s1 = (v1-1 > 0) ; v1 -= s1
        const _Float16* x0p = x0tile + (size_t)t * BB * HH;
        #pragma unroll
        for (int j = 0; j < 4; ++j) {
            int n = wave * 64 + j * 16 + lc;
            #pragma unroll
            for (int i = 0; i < 8; ++i) {
                int m = i + 8 * half;
                float x0 = (float)x0p[(size_t)m * HH + n];
                float v  = 0.9f * v1r[j * 8 + i] + x0;
                float s  = (v - 1.0f > 0.0f) ? 1.0f : 0.0f;
                v1r[j * 8 + i] = v - s;
                s1[m][n] = (_Float16)s;
            }
        }
        // prefetch next step's 16KB x0 tile (16 rows x 512 f16): one 64B chunk
        // per thread, hidden behind this step's GEMMs.
        if (t + 1 < TT) {
            const char* nxt = (const char*)(x0p + (size_t)BB * HH);
            __builtin_prefetch(nxt + tid * 64, 0, 1);
        }
        __syncthreads();

        // phase 2: x1 = s1 @ W1^T + b1 ; LIF on v2 ; spikes -> s2
        {
            v8f acc[4] = {};
            const _Float16* arow = &s1[lc][0];
            #pragma unroll 4
            for (int kt = 0; kt < 16; ++kt) {   // K = 512
                v16h a = load_afrag(arow, kt, half);
                #pragma unroll
                for (int j = 0; j < 4; ++j) {
                    v16h b = *(const v16h*)(bbase1 + (size_t)j * 16 * HH + kt * 32);
                    acc[j] = __builtin_amdgcn_wmma_f32_16x16x32_f16(false, a, false, b,
                                                                    (short)0, acc[j], false, false);
                }
            }
            #pragma unroll
            for (int j = 0; j < 4; ++j) {
                int n = wave * 64 + j * 16 + lc;
                #pragma unroll
                for (int i = 0; i < 8; ++i) {
                    int m = i + 8 * half;
                    float x1 = acc[j][i] + bb1[j];
                    float v  = 0.9f * v2r[j * 8 + i] + x1;
                    float s  = (v - 1.0f > 0.0f) ? 1.0f : 0.0f;
                    v2r[j * 8 + i] = v - s;
                    s2[m][n] = (_Float16)s;
                }
            }
        }
        __syncthreads();

        // phase 3 (waves 0,1 only; wave-uniform so EXEC is all-ones for WMMA):
        // x2 = s2 @ W2^T + b2 ; vo = 0.9 vo + x2
        if (wave < 2) {
            v8f c = {};
            const _Float16* arow2 = &s2[lc][0];
            #pragma unroll 4
            for (int kt = 0; kt < 16; ++kt) {
                v16h a = load_afrag(arow2, kt, half);
                v16h b = *(const v16h*)(bbase2 + kt * 32);
                c = __builtin_amdgcn_wmma_f32_16x16x32_f16(false, a, false, b,
                                                           (short)0, c, false, false);
            }
            #pragma unroll
            for (int i = 0; i < 8; ++i)
                vor[i] = 0.9f * vor[i] + (c[i] + bb2);
        }
        // no third barrier needed: step t+1 writes s1 (disjoint from s2/vo reads),
        // and s2 is only rewritten after the next phase-2 barrier which waves 0,1
        // reach only after finishing this GEMM.
    }

    if (wave < 2 && n2 < DOUT) {
        #pragma unroll
        for (int i = 0; i < 8; ++i) {
            int m = i + 8 * half;
            out[(size_t)(wg * 16 + m) * DOUT + n2] = vor[i];
        }
    }
}

extern "C" void kernel_launch(void* const* d_in, const int* in_sizes, int n_in,
                              void* d_out, int out_size, void* d_ws, size_t ws_size,
                              hipStream_t stream) {
    const float* inp  = (const float*)d_in[0];
    const float* W_in = (const float*)d_in[1];
    const float* b_in = (const float*)d_in[2];
    const float* W1   = (const float*)d_in[3];
    const float* b1   = (const float*)d_in[4];
    const float* W2   = (const float*)d_in[5];
    const float* b2   = (const float*)d_in[6];
    float* out = (float*)d_out;

    _Float16* w    = (_Float16*)d_ws;
    _Float16* Winf = w + OFF_WIN;
    _Float16* W1f  = w + OFF_W1;
    _Float16* W2f  = w + OFF_W2;
    _Float16* X0   = w + OFF_X0;

    // 638976 f16 weight elements == 2496 * 256
    cvt_weights_kernel<<<2496, 256, 0, stream>>>(W_in, W1, W2, w);
    // 128000 rows / 32 = 4000 M-tiles
    gemm0_kernel<<<4000, 256, 0, stream>>>(inp, b_in, Winf, X0);
    // 256 batch rows / 16 = 16 workgroups
    scan_kernel<<<16, 256, 0, stream>>>(b1, b2, W1f, W2f, X0, out);
}
